// CSNN_66907000537822
// MI455X (gfx1250) — matrix operand order repaired and checked by reading, not compiled, not run on hardware
//
#include <hip/hip_runtime.h>

typedef __attribute__((ext_vector_type(16))) _Float16 v16h;
typedef __attribute__((ext_vector_type(8)))  _Float16 v8h;
typedef __attribute__((ext_vector_type(8)))  float    v8f;

// ---------------------------------------------------------------------------
// Effective dense weight for a 3x3 SAME conv on a 3x3 grid, channels-major
// feature packing: f = ch*9 + (y*3+x).  K = input feature (plus bias row at
// K == IC*9), F = output feature.
// ---------------------------------------------------------------------------
__device__ __forceinline__ float conv_weff(const float* w, const float* b,
                                           int IC, int K, int F) {
  int co = F / 9, p = F % 9;
  int py = p / 3, px = p % 3;
  if (K < IC * 9) {
    int ci = K / 9, q = K % 9;
    int qy = q / 3, qx = q % 3;
    int ky = qy - py + 1, kx = qx - px + 1;
    if (ky < 0 || ky > 2 || kx < 0 || kx > 2) return 0.f;
    return w[(co * IC + ci) * 9 + ky * 3 + kx];
  }
  if (K == IC * 9) return b[co];   // folded bias (A supplies 1.0 at this K)
  return 0.f;
}

// ---------------------------------------------------------------------------
// Prep kernel, two jobs:
//  (a) 7 f16 B tiles (32x16) in WMMA B lane layout for the spike-driven
//      layers: element (K,N) at lane = (N&15) + 16*(K>=16), halfword = K&15.
//      Tile ids: 0: conv2 cols 0..15   1: conv2 cols 16..17
//                2: conv3 cols 0..15   3: conv3 cols 16..17
//                4: conv4 (9 cols)     5: fc1 (10 cols)     6: fc2 (2 cols)
//  (b) f32 effective conv1 matrix W1f[10][18] (row 9 = bias) for the exact
//      one-time conv1 (its input x is continuous -> keep it in f32).
// ---------------------------------------------------------------------------
__global__ void snn_prep_kernel(const float* __restrict__ w1, const float* __restrict__ b1,
                                const float* __restrict__ w2, const float* __restrict__ b2,
                                const float* __restrict__ w3, const float* __restrict__ b3,
                                const float* __restrict__ w4, const float* __restrict__ b4,
                                const float* __restrict__ wfc1, const float* __restrict__ wfc2,
                                _Float16* __restrict__ Bt, float* __restrict__ W1f) {
  int tid = blockIdx.x * blockDim.x + threadIdx.x;
  if (tid < 7 * 32) {
    int t = tid >> 5, lane = tid & 31;
    int n = lane & 15;
    for (int i = 0; i < 16; ++i) {
      int K = i + 16 * (lane >> 4);
      float v = 0.f;
      switch (t) {
        case 0: v = conv_weff(w2, b2, 2, K, n); break;
        case 1: v = (n < 2) ? conv_weff(w2, b2, 2, K, 16 + n) : 0.f; break;
        case 2: v = conv_weff(w3, b3, 2, K, n); break;
        case 3: v = (n < 2) ? conv_weff(w3, b3, 2, K, 16 + n) : 0.f; break;
        case 4: v = (n < 9) ? conv_weff(w4, b4, 2, K, n) : 0.f; break;
        case 5: v = (K < 9  && n < 10) ? wfc1[n * 9  + K] : 0.f; break;  // h @ wfc1.T
        case 6: v = (K < 10 && n < 2 ) ? wfc2[n * 10 + K] : 0.f; break;  // s5 @ wfc2.T
      }
      Bt[(t * 32 + lane) * 16 + i] = (_Float16)v;
    }
  } else if (tid - 7 * 32 < 10 * 18) {
    int idx = tid - 7 * 32;
    int K = idx / 18, F = idx % 18;
    W1f[K * 18 + F] = conv_weff(w1, b1, 1, K, F);   // K==9 -> bias row
  }
}

#define WMMA_F16(A, B, C)                                                     \
  __builtin_amdgcn_wmma_f32_16x16x32_f16(false, (A), false, (B), (short)0,    \
                                         (C), false, false)

// ---------------------------------------------------------------------------
// Main: one wave32 = 16 samples.  Membrane states in WMMA C/D layout (f32):
// lane l holds column N=l&15; VGPR g holds row M = g + 8*(l>=16).
// Spike transpose D->A via wave-private LDS row [16 samples][32 f16];
// slots 18..31 hold {1.0 (bias), 0...} and are written once.  Zero B rows
// make stale LDS feature slots harmless across the narrower layers.
// ---------------------------------------------------------------------------
__global__ __launch_bounds__(256) void snn_main_kernel(
    const float* __restrict__ x, const _Float16* __restrict__ Bt,
    const float* __restrict__ W1f, float* __restrict__ out, int Nsmp,
    const int* __restrict__ Tp) {
  __shared__ _Float16 smem[8][16][32];
  __shared__ float    c1sh[8][16][18];
  const int lane = threadIdx.x & 31;
  const int wid  = threadIdx.x >> 5;
  const int base = (blockIdx.x * 8 + wid) * 16;
  if (base >= Nsmp) return;                 // wave-uniform

  const int hi16 = lane >> 4;               // 0 or 1
  const int nl   = lane & 15;

  // Load the 7 B tiles into VGPRs (32 B per lane per tile).
  v16h B[7];
#pragma unroll
  for (int t = 0; t < 7; ++t)
    B[t] = *(const v16h*)(Bt + (t * 32 + lane) * 16);

  _Float16(&row_s)[16][32] = smem[wid];
  if (lane < 16) {
#pragma unroll
    for (int f = 18; f < 32; ++f)
      row_s[lane][f] = (_Float16)((f == 18) ? 1.f : 0.f);
  }

  // ---- exact f32 conv1 (input is time-invariant; done once) --------------
  // One sample per lane (lanes 0..15), 162 FMAs, then stage through LDS to
  // pick up the C/D accumulator layout.
  if (lane < 16) {
    int m = base + lane;
    if (m >= Nsmp) m = Nsmp - 1;
    const float* xm = x + m * 9;
    float xv[9];
#pragma unroll
    for (int k = 0; k < 9; ++k) xv[k] = xm[k];
#pragma unroll
    for (int f = 0; f < 18; ++f) {
      float s = W1f[9 * 18 + f];            // bias row
#pragma unroll
      for (int k = 0; k < 9; ++k) s += xv[k] * W1f[k * 18 + f];
      c1sh[wid][lane][f] = s;
    }
  }
  v8f c1a, c1b;
#pragma unroll
  for (int g = 0; g < 8; ++g) {
    int M = g + 8 * hi16;
    c1a[g] = c1sh[wid][M][nl];
    c1b[g] = (nl < 2) ? c1sh[wid][M][16 + nl] : 0.f;
  }

  const v8f zero = {0.f, 0.f, 0.f, 0.f, 0.f, 0.f, 0.f, 0.f};
  v8f v1a = zero, v1b = zero, v2a = zero, v2b = zero, v3a = zero, v3b = zero;
  v8f v4 = zero, v5 = zero, v6 = zero, acc = zero;

  // IF update on one accumulator tile; store spikes into LDS column `col`
  // when col < colLim (keeps bias/zero slots 18..31 intact).
  auto if_tile = [&](v8f& v, const v8f& u, float thr, int col, int colLim) {
#pragma unroll
    for (int g = 0; g < 8; ++g) {
      float nv = v[g] + u[g];
      bool  f  = nv >= thr;
      if (col < colLim)
        row_s[g + 8 * hi16][col] = (_Float16)(f ? 1.f : 0.f);
      v[g] = f ? 0.f : nv;
    }
  };
  // Gather next-layer A from the spike row (two 16B LDS reads per lane).
  // A layout: lane holds row M=nl; halfword i -> K = i + 16*hi16.
  v16h a;
  auto load_a = [&]() {
    const _Float16* r = &row_s[nl][0];
    int off = 16 * hi16;
    v8h lo = *(const v8h*)(r + off);
    v8h hi = *(const v8h*)(r + 8 + off);
    a = __builtin_shufflevector(lo, hi, 0, 1, 2, 3, 4, 5, 6, 7, 8, 9, 10, 11,
                                12, 13, 14, 15);
  };

  const int T = *Tp;
  for (int t = 0; t < T; ++t) {
    // layer 1 (thr 20): integrate precomputed c1
    if_tile(v1a, c1a, 20.f, nl, 16);
    if_tile(v1b, c1b, 20.f, 16 + nl, 18);
    // layer 2 (conv2, thr 10)
    load_a();
    v8f u2a = WMMA_F16(a, B[0], zero);
    v8f u2b = WMMA_F16(a, B[1], zero);
    if_tile(v2a, u2a, 10.f, nl, 16);
    if_tile(v2b, u2b, 10.f, 16 + nl, 18);
    // layer 3 (conv3, thr 8)
    load_a();
    v8f u3a = WMMA_F16(a, B[2], zero);
    v8f u3b = WMMA_F16(a, B[3], zero);
    if_tile(v3a, u3a, 8.f, nl, 16);
    if_tile(v3b, u3b, 8.f, 16 + nl, 18);
    // layer 4 (conv4 -> 9 feats, thr 8); cols 9..15 write 0 (harmless)
    load_a();
    v8f u4 = WMMA_F16(a, B[4], zero);
    if_tile(v4, u4, 8.f, nl, 16);
    // fc1 (-> 10 feats, thr 30)
    load_a();
    v8f u5 = WMMA_F16(a, B[5], zero);
    if_tile(v5, u5, 30.f, nl, 16);
    // fc2 (-> 2 feats, thr 30): accumulate output spikes
    load_a();
    v8f u6 = WMMA_F16(a, B[6], zero);
#pragma unroll
    for (int g = 0; g < 8; ++g) {
      float nv = v6[g] + u6[g];
      bool  f  = nv >= 30.f;
      acc[g] += f ? 1.f : 0.f;
      v6[g] = f ? 0.f : nv;
    }
  }

  // firing rate out[m][0..1]; D layout: lane nl = output index, VGPR g = sample
  float invT = 1.f / (float)T;
  if (nl < 2) {
#pragma unroll
    for (int g = 0; g < 8; ++g) {
      int m = base + g + 8 * hi16;
      if (m < Nsmp) out[m * 2 + nl] = acc[g] * invT;
    }
  }
}

extern "C" void kernel_launch(void* const* d_in, const int* in_sizes, int n_in,
                              void* d_out, int out_size, void* d_ws, size_t ws_size,
                              hipStream_t stream) {
  const float* x    = (const float*)d_in[0];
  const float* w1   = (const float*)d_in[1];
  const float* b1   = (const float*)d_in[2];
  const float* w2   = (const float*)d_in[3];
  const float* b2   = (const float*)d_in[4];
  const float* w3   = (const float*)d_in[5];
  const float* b3   = (const float*)d_in[6];
  const float* w4   = (const float*)d_in[7];
  const float* b4   = (const float*)d_in[8];
  const float* wfc1 = (const float*)d_in[9];
  const float* wfc2 = (const float*)d_in[10];
  const int*   Tp   = (const int*)d_in[11];

  int N = in_sizes[0] / 9;                         // x is [N,1,3,3]
  _Float16* Bt  = (_Float16*)d_ws;                 // 7 tiles * 512 f16 = 7 KB
  float*    W1f = (float*)((char*)d_ws + 7 * 512 * sizeof(_Float16));  // 720 B

  snn_prep_kernel<<<1, 512, 0, stream>>>(w1, b1, w2, b2, w3, b3, w4, b4,
                                         wfc1, wfc2, Bt, W1f);

  int waves  = (N + 15) / 16;
  int blocks = (waves + 7) / 8;
  snn_main_kernel<<<blocks, 256, 0, stream>>>(x, Bt, W1f, (float*)d_out, N, Tp);
}
